// AttentionDecoderLSTM_41790031790661
// MI455X (gfx1250) — compile-verified, matrix operand order checked by reading, not run on hardware
//
#include <hip/hip_runtime.h>
#include <hip/hip_bf16.h>

// ---------------------------------------------------------------------------
// Problem constants
// ---------------------------------------------------------------------------
constexpr int Bc = 64;      // batch
constexpr int Sc = 2048;    // source length
constexpr int Hc = 512;     // hidden
constexpr int Ec = 1024;    // encoder dim
constexpr int Dc = 256;     // embed dim
constexpr int Vc = 50257;   // vocab
constexpr int XKc = Dc + Ec;        // 1280 : LSTM input concat
constexpr int PKc = Hc + Ec + Dc;   // 1792 : prediction input concat
constexpr int VcFull = (Vc / 128) * 128;   // 50176 : full 128-col blocks of V

typedef __attribute__((ext_vector_type(16))) __bf16    v16bf;
typedef __attribute__((ext_vector_type(8)))  __bf16    v8bf;
typedef __attribute__((ext_vector_type(8)))  float     v8f;
typedef __attribute__((ext_vector_type(8)))  unsigned  v8u;

// ---------------------------------------------------------------------------
// Packed fp32x2 -> bf16x2 (RNE). Prefers v_cvt_pk_bf16_f32 when available.
// ---------------------------------------------------------------------------
__device__ __forceinline__ unsigned pk2(float x, float y) {
#if __has_builtin(__builtin_amdgcn_cvt_pk_bf16_f32)
  typedef __attribute__((ext_vector_type(2))) __bf16 v2bf;
  v2bf r = __builtin_amdgcn_cvt_pk_bf16_f32(x, y);
  return __builtin_bit_cast(unsigned, r);
#else
  unsigned ux = __builtin_bit_cast(unsigned, x);
  ux += 0x7FFFu + ((ux >> 16) & 1u);
  unsigned uy = __builtin_bit_cast(unsigned, y);
  uy += 0x7FFFu + ((uy >> 16) & 1u);
  return (ux >> 16) | (uy & 0xFFFF0000u);
#endif
}

__device__ __forceinline__ v16bf zero_v16bf() {
  v8u z = {0u,0u,0u,0u,0u,0u,0u,0u};
  return __builtin_bit_cast(v16bf, z);
}

// ---------------------------------------------------------------------------
// CDNA5 16-bit A/B fragment layout (wave32), 16x32 tile:
//   lanes 0..15 : row = rbase+lane,     K = kbase+{0..7, 16..23}
//   lanes 16..31: row = rbase+lane-16,  K = kbase+{8..15, 24..31}
// ---------------------------------------------------------------------------

// fp32 row-major, in-flight packed conversion, NO row guard (rows known valid)
__device__ __forceinline__ v16bf load_frag_f32cvt_nog(const float* __restrict__ M,
                                                      int ld, int rbase, int kbase) {
  const int lane = threadIdx.x & 31;
  const int r = rbase + (lane & 15);
  const int k = kbase + ((lane >> 4) << 3);
  const float4* q = reinterpret_cast<const float4*>(M + (size_t)r * ld + k);
  float4 x0 = q[0];
  float4 x1 = q[1];
  float4 x2 = q[4];   // +16 floats
  float4 x3 = q[5];
  v8u u;
  u[0] = pk2(x0.x, x0.y); u[1] = pk2(x0.z, x0.w);
  u[2] = pk2(x1.x, x1.y); u[3] = pk2(x1.z, x1.w);
  u[4] = pk2(x2.x, x2.y); u[5] = pk2(x2.z, x2.w);
  u[6] = pk2(x3.x, x3.y); u[7] = pk2(x3.z, x3.w);
  return __builtin_bit_cast(v16bf, u);
}

// fp32 row-major with row guard (ragged-N W operand only)
__device__ __forceinline__ v16bf load_frag_f32cvt(const float* __restrict__ M,
                                                  int ld, int rbase, int kbase,
                                                  int rmax) {
  const int lane = threadIdx.x & 31;
  const int r = rbase + (lane & 15);
  if (r < rmax) return load_frag_f32cvt_nog(M, ld, rbase, kbase);
  return zero_v16bf();
}

// bf16 row-major (global or LDS): two raw 16-byte loads, no conversion, no guard
__device__ __forceinline__ v16bf load_frag_b16_nog(const __bf16* __restrict__ M,
                                                   int ld, int rbase, int kbase) {
  const int lane = threadIdx.x & 31;
  const int r = rbase + (lane & 15);
  const int k = kbase + ((lane >> 4) << 3);
  const v8bf* p = reinterpret_cast<const v8bf*>(M + (size_t)r * ld + k);
  v8bf lo = p[0];
  v8bf hi = p[2];   // +16 elements
  return __builtin_shufflevector(lo, hi, 0,1,2,3,4,5,6,7,8,9,10,11,12,13,14,15);
}

// ---------------------------------------------------------------------------
// fp32 -> bf16 bulk conversion (weights pre-pass). n2 = element-pairs.
// ---------------------------------------------------------------------------
__global__ __launch_bounds__(256)
void cvt_f32_bf16_kernel(const float* __restrict__ src,
                         __bf16* __restrict__ dst, int n2) {
  const int i = blockIdx.x * 256 + threadIdx.x;
  if (i < n2) {
    const float2 v = reinterpret_cast<const float2*>(src)[i];
    reinterpret_cast<unsigned*>(dst)[i] = pk2(v.x, v.y);
  }
}

// ---------------------------------------------------------------------------
// C[64, n0 + 128*gridX cols of N] = A[64,K] . W[N,K]^T (+ bias[N]) (+= if acc)
// One launch covers ALL of M=64 per column-block so streamed W is read from
// HBM exactly once. 256 threads = 8 waves: wave_m = wave&3 (4 m-tiles),
// wave_n = wave>>2 (2 n-groups of 64). Block tile 64(M) x 128(N). K % 32 == 0.
// WB16 : W pre-converted bf16 (Wb) vs fp32 (Wf, in-flight packed cvt).
// NGUARD: predicate W loads / C accesses on n < N (ragged tail block only).
// n0   : column offset of this launch (tail-split for ragged N).
// N is also the row stride of C and row count bound of W.
// ---------------------------------------------------------------------------
template <bool WB16, bool NGUARD>
__global__ __launch_bounds__(256)
void wmma_gemm64(const float* __restrict__ A,
                 const float* __restrict__ Wf,
                 const __bf16* __restrict__ Wb,
                 const float* __restrict__ bias,
                 float* __restrict__ C,
                 int N, int K, int accumulate, int n0) {
  const int wave = threadIdx.x >> 5;
  const int lane = threadIdx.x & 31;
  const int mbase = (wave & 3) * 16;
  const int nbase = n0 + blockIdx.x * 128 + (wave >> 2) * 64;
  const int ncol  = lane & 15;
  const int rrow  = (lane >> 4) << 3;

  v8f acc[4];
  if (accumulate) {
#pragma unroll
    for (int t = 0; t < 4; ++t) {
      const int n = nbase + 16 * t + ncol;
#pragma unroll
      for (int r = 0; r < 8; ++r) {
        const int m = mbase + rrow + r;
        acc[t][r] = (!NGUARD || n < N) ? C[(size_t)m * N + n] : 0.f;
      }
    }
  } else {
#pragma unroll
    for (int t = 0; t < 4; ++t) acc[t] = (v8f){0.f,0.f,0.f,0.f,0.f,0.f,0.f,0.f};
  }

  for (int kb = 0; kb < K; kb += 32) {
    if (kb + 32 < K) {   // prefetch the streamed W operand -> global_prefetch_b8
      if (WB16) __builtin_prefetch(Wb + (size_t)(nbase + ncol) * K + kb + 32, 0, 3);
      else      __builtin_prefetch(Wf + (size_t)(nbase + ncol) * K + kb + 32, 0, 3);
    }
    // Preload all W fragments of this k-step, then run the WMMA chain so the
    // scheduler can retire s_wait_loadcnt incrementally instead of 0 per WMMA.
    v16bf wfrag[4];
#pragma unroll
    for (int t = 0; t < 4; ++t) {
      if (WB16)        wfrag[t] = load_frag_b16_nog(Wb, K, nbase + 16 * t, kb);
      else if (NGUARD) wfrag[t] = load_frag_f32cvt(Wf, K, nbase + 16 * t, kb, N);
      else             wfrag[t] = load_frag_f32cvt_nog(Wf, K, nbase + 16 * t, kb);
    }
    const v16bf a = load_frag_f32cvt_nog(A, K, mbase, kb);   // M = 64, always valid
#pragma unroll
    for (int t = 0; t < 4; ++t)
      acc[t] = __builtin_amdgcn_wmma_f32_16x16x32_bf16(
          false, a, false, wfrag[t], (short)0, acc[t], false, false);
  }

#pragma unroll
  for (int t = 0; t < 4; ++t) {
    const int n = nbase + 16 * t + ncol;
    if (!NGUARD || n < N) {
      const float bv = bias ? bias[n] : 0.f;
#pragma unroll
      for (int r = 0; r < 8; ++r) {
        const int m = mbase + rrow + r;
        C[(size_t)m * N + n] = acc[t][r] + bv;
      }
    }
  }
}

// ---------------------------------------------------------------------------
// Fused attention-score kernel:
//   scores[b,s] = sum_h W_v[h]*tanh(hid_proj[b,h]+b_ae[h]+sum_e enc[b,s,e]W_ae[h,e])
// Block = (s-tile of 32, b); 8 waves; wave tile 16(s) x 128(h).
// enc tile (32x1024) is packed-converted once into LDS as bf16; A-fragments
// come from ds_load_b128. W_ae is pre-converted bf16 in global ws.
// ---------------------------------------------------------------------------
constexpr int LDA = Ec + 8;   // +8 bf16 (16B) row pad: conflict-spread, 16B-aligned

__global__ __launch_bounds__(256)
void attn_scores_kernel(const float* __restrict__ enc,
                        const __bf16* __restrict__ W_aeb,
                        const float* __restrict__ hidproj,  // B*H, incl. b_ah
                        const float* __restrict__ b_ae,
                        const float* __restrict__ W_v,
                        const int*   __restrict__ mask,
                        float* __restrict__ scores) {
  __shared__ __align__(16) __bf16 tileA[32 * LDA];
  __shared__ float rowsum[32];

  const int b    = blockIdx.y;
  const int s0   = blockIdx.x * 32;
  const int tid  = threadIdx.x;
  const int wave = tid >> 5;
  const int lane = tid & 31;
  const int mbase = (wave >> 2) * 16;   // local s-row base
  const int nbase = (wave & 3) * 128;   // h base: 8 tiles of 16

  // --- cooperative fp32 -> bf16 staging of the 32x1024 enc tile ------------
  const float* Ab = enc + ((size_t)b * Sc + s0) * Ec;
  for (int c = tid; c < (32 * Ec) / 8; c += 256) {   // 8-elt chunks
    const int row = c >> 7;            // Ec/8 == 128 chunks per row
    const int col = (c & 127) * 8;
    const float4* q = reinterpret_cast<const float4*>(Ab + (size_t)row * Ec + col);
    const float4 a0 = q[0];
    const float4 a1 = q[1];
    uint4 u;
    u.x = pk2(a0.x, a0.y); u.y = pk2(a0.z, a0.w);
    u.z = pk2(a1.x, a1.y); u.w = pk2(a1.z, a1.w);
    *reinterpret_cast<uint4*>(&tileA[row * LDA + col]) = u;
  }
  if (tid < 32) rowsum[tid] = 0.f;
  __syncthreads();

  v8f acc[8];
#pragma unroll
  for (int t = 0; t < 8; ++t) acc[t] = (v8f){0.f,0.f,0.f,0.f,0.f,0.f,0.f,0.f};

  for (int kb = 0; kb < Ec; kb += 32) {
    // Preload all 8 W fragments, then the A fragment, then the WMMA chain.
    v16bf wfrag[8];
#pragma unroll
    for (int t = 0; t < 8; ++t)
      wfrag[t] = load_frag_b16_nog(W_aeb, Ec, nbase + 16 * t, kb);
    const v16bf a = load_frag_b16_nog(tileA, LDA, mbase, kb);   // ds_load_b128
#pragma unroll
    for (int t = 0; t < 8; ++t)
      acc[t] = __builtin_amdgcn_wmma_f32_16x16x32_bf16(
          false, a, false, wfrag[t], (short)0, acc[t], false, false);
  }

  // --- epilogue: tanh + dot with W_v, reduce over h ------------------------
  float part[8];
#pragma unroll
  for (int r = 0; r < 8; ++r) part[r] = 0.f;

#pragma unroll
  for (int t = 0; t < 8; ++t) {
    const int h  = nbase + 16 * t + (lane & 15);
    const float hb = hidproj[b * Hc + h] + b_ae[h];
    const float wv = W_v[h];
#pragma unroll
    for (int r = 0; r < 8; ++r)
      part[r] += tanhf(acc[t][r] + hb) * wv;
  }

#pragma unroll
  for (int off = 1; off < 16; off <<= 1) {   // xor 1,2,4,8 stays within half
#pragma unroll
    for (int r = 0; r < 8; ++r)
      part[r] += __shfl_xor(part[r], off, 32);
  }
  if ((lane & 15) == 0) {
    const int rowbase = mbase + ((lane >> 4) << 3);
#pragma unroll
    for (int r = 0; r < 8; ++r)
      atomicAdd(&rowsum[rowbase + r], part[r]);   // ds_add_f32
  }
  __syncthreads();

  if (tid < 32) {
    const int s = s0 + tid;
    const float v = rowsum[tid];
    scores[(size_t)b * Sc + s] = (mask[(size_t)b * Sc + s] == 0) ? -1e10f : v;
  }
}

// ---------------------------------------------------------------------------
__global__ __launch_bounds__(256)
void softmax_kernel(const float* __restrict__ scores,
                    float* __restrict__ attn_ws,
                    float* __restrict__ attn_out) {
  const int b = blockIdx.x;
  const int tid = threadIdx.x;
  const int wave = tid >> 5, lane = tid & 31;
  __shared__ float red[8];
  __shared__ float s_max, s_sum;

  float m = -3.4e38f;
  for (int s = tid; s < Sc; s += 256) m = fmaxf(m, scores[(size_t)b * Sc + s]);
#pragma unroll
  for (int off = 1; off < 32; off <<= 1) m = fmaxf(m, __shfl_xor(m, off, 32));
  if (lane == 0) red[wave] = m;
  __syncthreads();
  if (tid == 0) {
    float mm = red[0];
#pragma unroll
    for (int i = 1; i < 8; ++i) mm = fmaxf(mm, red[i]);
    s_max = mm;
  }
  __syncthreads();
  const float mm = s_max;

  float acc = 0.f;
  for (int s = tid; s < Sc; s += 256) {
    const float e = __expf(scores[(size_t)b * Sc + s] - mm);
    attn_ws[(size_t)b * Sc + s] = e;
    acc += e;
  }
#pragma unroll
  for (int off = 1; off < 32; off <<= 1) acc += __shfl_xor(acc, off, 32);
  if (lane == 0) red[wave] = acc;
  __syncthreads();
  if (tid == 0) {
    float ss = 0.f;
#pragma unroll
    for (int i = 0; i < 8; ++i) ss += red[i];
    s_sum = ss;
  }
  __syncthreads();
  const float inv = 1.f / s_sum;
  for (int s = tid; s < Sc; s += 256) {
    const float v = attn_ws[(size_t)b * Sc + s] * inv;
    attn_ws[(size_t)b * Sc + s]  = v;
    attn_out[(size_t)b * Sc + s] = v;
  }
}

// context[b,e] = sum_s attn[b,s] * enc[b,s,e]
__global__ __launch_bounds__(256)
void context_kernel(const float* __restrict__ enc,
                    const float* __restrict__ attn,
                    float* __restrict__ context) {
  const int b = blockIdx.y;
  const int e = blockIdx.x * 256 + threadIdx.x;
  const float* p = enc + (size_t)b * Sc * Ec + e;
  const float* a = attn + (size_t)b * Sc;
  float acc = 0.f;
#pragma unroll 8
  for (int s = 0; s < Sc; ++s) acc += a[s] * p[(size_t)s * Ec];
  context[(size_t)b * Ec + e] = acc;
}

__global__ void embed_kernel(const int* __restrict__ token,
                             const float* __restrict__ table,
                             float* __restrict__ emb) {
  const int b = blockIdx.x;
  const int d = threadIdx.x;            // D == 256 == blockDim
  emb[(size_t)b * Dc + d] = table[(size_t)token[b] * Dc + d];
}

__global__ void concat_x_kernel(const float* __restrict__ emb,
                                const float* __restrict__ ctx,
                                float* __restrict__ x) {
  const int b = blockIdx.x;
  for (int i = threadIdx.x; i < XKc; i += 256)
    x[(size_t)b * XKc + i] = (i < Dc) ? emb[(size_t)b * Dc + i]
                                      : ctx[(size_t)b * Ec + (i - Dc)];
}

__global__ void lstm_kernel(const float* __restrict__ gates,
                            const float* __restrict__ c0,
                            float* __restrict__ h1_out,
                            float* __restrict__ c1_out,
                            float* __restrict__ predin) {
  const int idx = blockIdx.x * 256 + threadIdx.x;   // b*H + h
  if (idx >= Bc * Hc) return;
  const int b = idx / Hc, h = idx % Hc;
  const float* g = gates + (size_t)b * 4 * Hc;
  float i = g[h], f = g[Hc + h], gg = g[2 * Hc + h], o = g[3 * Hc + h];
  i = 1.f / (1.f + __expf(-i));
  f = 1.f / (1.f + __expf(-f));
  o = 1.f / (1.f + __expf(-o));
  gg = tanhf(gg);
  const float c1 = f * c0[idx] + i * gg;
  const float h1 = o * tanhf(c1);
  h1_out[idx] = h1;
  c1_out[idx] = c1;
  predin[(size_t)b * PKc + h] = h1;
}

__global__ void concat_pred_kernel(const float* __restrict__ ctx,
                                   const float* __restrict__ emb,
                                   float* __restrict__ predin) {
  const int b = blockIdx.x;
  for (int i = threadIdx.x; i < Ec + Dc; i += 256)
    predin[(size_t)b * PKc + Hc + i] =
        (i < Ec) ? ctx[(size_t)b * Ec + i] : emb[(size_t)b * Dc + (i - Ec)];
}

// ---------------------------------------------------------------------------
extern "C" void kernel_launch(void* const* d_in, const int* in_sizes, int n_in,
                              void* d_out, int out_size, void* d_ws, size_t ws_size,
                              hipStream_t stream) {
  (void)in_sizes; (void)n_in; (void)out_size; (void)ws_size;

  const int*   input_token = (const int*)  d_in[0];
  const float* hidden      = (const float*)d_in[1];   // (1,B,H) -> h0
  const float* cell        = (const float*)d_in[2];   // (1,B,H) -> c0
  const float* enc         = (const float*)d_in[3];   // (B,S,E)
  const int*   mask        = (const int*)  d_in[4];
  const float* emb_table   = (const float*)d_in[5];
  const float* W_ah        = (const float*)d_in[6];
  const float* b_ah        = (const float*)d_in[7];
  const float* W_ae        = (const float*)d_in[8];
  const float* b_ae        = (const float*)d_in[9];
  const float* W_v         = (const float*)d_in[10];
  const float* W_ih        = (const float*)d_in[11];
  const float* W_hh        = (const float*)d_in[12];
  const float* b_ih        = (const float*)d_in[13];
  const float* b_hh        = (const float*)d_in[14];
  const float* W_out       = (const float*)d_in[15];
  const float* b_out       = (const float*)d_in[16];

  // outputs: prediction | h1 | c1 | attn_w
  float* out      = (float*)d_out;
  float* pred     = out;
  float* h1_out   = out + (size_t)Bc * Vc;
  float* c1_out   = h1_out + (size_t)Bc * Hc;
  float* attn_out = c1_out + (size_t)Bc * Hc;

  // workspace: fp32 region then bf16 region
  float* ws        = (float*)d_ws;
  float* w_emb     = ws;                              // B*D
  float* w_hidproj = w_emb     + (size_t)Bc * Dc;     // B*H
  float* w_scores  = w_hidproj + (size_t)Bc * Hc;     // B*S
  float* w_attn    = w_scores  + (size_t)Bc * Sc;     // B*S
  float* w_ctx     = w_attn    + (size_t)Bc * Sc;     // B*E
  float* w_x       = w_ctx     + (size_t)Bc * Ec;     // B*XK
  float* w_gates   = w_x       + (size_t)Bc * XKc;    // B*4H
  float* w_predin  = w_gates   + (size_t)Bc * 4 * Hc; // B*PK
  __bf16* wb_base  = (__bf16*)(w_predin + (size_t)Bc * PKc);  // 16B-aligned
  __bf16* wb_ae    = wb_base;                          // H*E
  __bf16* wb_ih    = wb_ae + (size_t)Hc * Ec;          // 4H*XK
  __bf16* wb_hh    = wb_ih + (size_t)4 * Hc * XKc;     // 4H*H

  // --- weight pre-conversion to bf16 (reused, L2-resident) -----------------
  {
    const int n2a = Hc * Ec / 2;
    const int n2i = 4 * Hc * XKc / 2;
    const int n2h = 4 * Hc * Hc / 2;
    cvt_f32_bf16_kernel<<<dim3((n2a + 255) / 256), dim3(256), 0, stream>>>(W_ae, wb_ae, n2a);
    cvt_f32_bf16_kernel<<<dim3((n2i + 255) / 256), dim3(256), 0, stream>>>(W_ih, wb_ih, n2i);
    cvt_f32_bf16_kernel<<<dim3((n2h + 255) / 256), dim3(256), 0, stream>>>(W_hh, wb_hh, n2h);
  }

  embed_kernel<<<dim3(Bc), dim3(Dc), 0, stream>>>(input_token, emb_table, w_emb);

  // hid_proj = h0 @ W_ah^T + b_ah   (tiny: fp32 W in-flight, N=512 no guard)
  wmma_gemm64<false, false><<<dim3(Hc / 128), dim3(256), 0, stream>>>(
      hidden, W_ah, nullptr, b_ah, w_hidproj, Hc, Hc, 0, 0);

  // fused enc_proj -> tanh -> .W_v -> masked scores (137 GFLOP, HBM-bound)
  attn_scores_kernel<<<dim3(Sc / 32, Bc), dim3(256), 0, stream>>>(
      enc, wb_ae, w_hidproj, b_ae, W_v, mask, w_scores);

  softmax_kernel<<<dim3(Bc), dim3(256), 0, stream>>>(w_scores, w_attn, attn_out);

  context_kernel<<<dim3(Ec / 256, Bc), dim3(256), 0, stream>>>(enc, w_attn, w_ctx);

  concat_x_kernel<<<dim3(Bc), dim3(256), 0, stream>>>(w_emb, w_ctx, w_x);

  // gates = x @ W_ih^T + b_ih ; then += h0 @ W_hh^T + b_hh (bf16 weights)
  wmma_gemm64<true, false><<<dim3(4 * Hc / 128), dim3(256), 0, stream>>>(
      w_x, nullptr, wb_ih, b_ih, w_gates, 4 * Hc, XKc, 0, 0);
  wmma_gemm64<true, false><<<dim3(4 * Hc / 128), dim3(256), 0, stream>>>(
      hidden, nullptr, wb_hh, b_hh, w_gates, 4 * Hc, Hc, 1, 0);

  lstm_kernel<<<dim3((Bc * Hc + 255) / 256), dim3(256), 0, stream>>>(
      w_gates, cell, h1_out, c1_out, w_predin);

  concat_pred_kernel<<<dim3(Bc), dim3(256), 0, stream>>>(w_ctx, w_emb, w_predin);

  // prediction = pred_in @ W_out^T + b_out (W_out streamed once).
  // Tail-split so 392 full blocks run the unguarded inner loop; only the
  // single ragged block (81 cols) pays the n<N predication.
  wmma_gemm64<false, false><<<dim3(VcFull / 128), dim3(256), 0, stream>>>(
      w_predin, W_out, nullptr, b_out, pred, Vc, PKc, 0, 0);
  wmma_gemm64<false, true><<<dim3(1), dim3(256), 0, stream>>>(
      w_predin, W_out, nullptr, b_out, pred, Vc, PKc, 0, VcFull);
}
